// GlobalLSTM_29317446763025
// MI455X (gfx1250) — matrix operand order repaired and checked by reading, not compiled, not run on hardware
//
#include <hip/hip_runtime.h>

typedef __attribute__((ext_vector_type(16))) _Float16 v16h;
typedef __attribute__((ext_vector_type(2)))  _Float16 v2h;
typedef __attribute__((ext_vector_type(8)))  float    v8f;
typedef __attribute__((ext_vector_type(2)))  float    v2f;

#define SWZ_SWAPX16 0x401F  // group-of-32: xor=0x10, or=0, and=0x1F

#if __has_builtin(__builtin_amdgcn_tanhf)
// Native CDNA5 V_TANH_F32 (trans unit, co-executes with XDL WMMA)
__device__ __forceinline__ float fast_tanh(float v) {
    return __builtin_amdgcn_tanhf(v);
}
// sigmoid(2*hg) given pre-halved argument hg = g/2 (0.5 folded into weights)
__device__ __forceinline__ float sig_from_half(float hg) {
    return __builtin_fmaf(0.5f, __builtin_amdgcn_tanhf(hg), 0.5f);
}
#else
__device__ __forceinline__ float fast_tanh(float v) {
    return 2.0f * __builtin_amdgcn_rcpf(1.0f + __builtin_amdgcn_exp2f(-2.8853900817779268f * v)) - 1.0f;
}
__device__ __forceinline__ float sig_from_half(float hg) {
    // sigmoid(2*hg) = 1 / (1 + 2^(-2*hg*log2(e)))
    return __builtin_amdgcn_rcpf(1.0f + __builtin_amdgcn_exp2f(-2.8853900817779268f * hg));
}
#endif

// One wave handles 16 batch rows. Per timestep:
//   G^T[128,16] = [W_ih | b] @ [x_t;1]^T   (8x V_WMMA_F32_16X16X4_F32, fp32)
//               +  W_hh      @ h^T         (8x V_WMMA_F32_16X16X32_F16, K=32=H)
// Gate tiles (M): 0,1 = i ; 2,3 = f ; 4,5 = g ; 6,7 = o.
// Weights of i/f/o tiles are pre-scaled by 0.5 so sigmoid = fma(0.5,tanh(.),0.5)
// with no per-step multiply. h feedback rebuilt into f16 B-layout with
// v_cvt_pk only: each lane's D rows already match B's k order (no lane swap, no LDS traffic).
__global__ __launch_bounds__(256)
void GlobalLSTM_kernel(const float* __restrict__ x,
                       const int*   __restrict__ mat_idx,
                       const int*   __restrict__ freq_idx,
                       const float* __restrict__ W_ih,
                       const float* __restrict__ W_hh,
                       const float* __restrict__ b_ih,
                       const float* __restrict__ b_hh,
                       const float* __restrict__ mat_emb,
                       const float* __restrict__ freq_emb,
                       const float* __restrict__ fc_w,
                       const float* __restrict__ fc_b,
                       float* __restrict__ out,
                       int B, int T) {
    const int lane = threadIdx.x & 31;
    const int wv   = blockIdx.x * (blockDim.x >> 5) + (threadIdx.x >> 5);
    const int b0   = wv * 16;
    if (b0 >= B) return;                 // wave-uniform guard
    const int r16 = lane & 15;           // column n (batch within tile) / A row
    const int lh  = lane >> 4;           // lane half
    const int kb  = lh * 8;              // A-layout K block base

    // ---- resident A1 tiles: W_hh [128,32] as 8 f16 16x32 A-tiles ----
    // i/f/o gate tiles (0,1,2,3,6,7) pre-scaled by 0.5 for mul-free sigmoid.
    v16h a1[8];
#pragma unroll
    for (int t = 0; t < 8; ++t) {
        const float sc = (t == 4 || t == 5) ? 1.0f : 0.5f;
        const float* wr = W_hh + (t * 16 + r16) * 32;
#pragma unroll
        for (int e = 0; e < 8; ++e) {
            a1[t][e]     = (_Float16)(sc * wr[kb + e]);        // K = kb .. kb+7
            a1[t][e + 8] = (_Float16)(sc * wr[kb + 16 + e]);   // K = kb+16 .. kb+23
        }
    }

    // ---- resident A2 tiles: [W_ih | (b_ih+b_hh)] [128,4] as 8 fp32 16x4 A-tiles ----
    v2f a2[8];
#pragma unroll
    for (int t = 0; t < 8; ++t) {
        const float sc = (t == 4 || t == 5) ? 1.0f : 0.5f;
        const int m = t * 16 + r16;
#pragma unroll
        for (int j = 0; j < 2; ++j) {
            const int K = 2 * lh + j;    // lanes 0-15: K=0,1 ; lanes 16-31: K=2,3
            a2[t][j] = sc * ((K < 3) ? W_ih[m * 3 + K] : (b_ih[m] + b_hh[m]));
        }
    }

    v8f vzero;
    v16h hb;                              // h^T in f16 B-layout (starts at 0)
#pragma unroll
    for (int e = 0; e < 8; ++e)  vzero[e] = 0.0f;
#pragma unroll
    for (int e = 0; e < 16; ++e) hb[e] = (_Float16)0.0f;

    float c0[8], c1[8], h0f[8], h1f[8];   // cell / hidden state, D-layout tiles 0,1
#pragma unroll
    for (int r = 0; r < 8; ++r) { c0[r] = c1[r] = h0f[r] = h1f[r] = 0.0f; }

    const float* px = x + (long)(b0 + r16) * (long)T * 3;
    const float* qx = px + 2 * lh;        // half0 -> &x[b,t,0], half1 -> &x[b,t,2]

#pragma unroll 1
    for (int t = 0; t < T; ++t) {
        // per-lane contiguous stream of this batch row's x_t (2 loads/lane)
        const float xa  = qx[t * 3];          // half0: x0 ; half1: x2
        const float x1v = px[t * 3 + 1];      // used by half0 only
        if (t + 32 < T) __builtin_prefetch(px + (t + 32) * 3, 0, 1);

        v2f bx;                               // [x_t;1] as fp32 4x16 B-matrix
        bx[0] = xa;                           // K=0 (half0) / K=2 (half1)
        bx[1] = lh ? 1.0f : x1v;              // K=1 (half0) / K=3 (half1, bias row)

        v8f g[8];
#pragma unroll
        for (int q = 0; q < 8; ++q) {
            g[q] = __builtin_amdgcn_wmma_f32_16x16x4_f32(
                false, a2[q], false, bx, (short)0, vzero, false, false);
            g[q] = __builtin_amdgcn_wmma_f32_16x16x32_f16(
                false, a1[q], false, hb, (short)0, g[q], false, false);
        }

        // activations + state update (i/f/o tiles hold g/2; tanh+fma only)
#pragma unroll
        for (int r = 0; r < 8; ++r) {
            const float i0 = sig_from_half(g[0][r]);
            const float f0 = sig_from_half(g[2][r]);
            const float g0 = fast_tanh    (g[4][r]);
            const float o0 = sig_from_half(g[6][r]);
            c0[r]  = __builtin_fmaf(f0, c0[r], i0 * g0);
            h0f[r] = o0 * fast_tanh(c0[r]);

            const float i1 = sig_from_half(g[1][r]);
            const float f1 = sig_from_half(g[3][r]);
            const float g1 = fast_tanh    (g[5][r]);
            const float o1 = sig_from_half(g[7][r]);
            c1[r]  = __builtin_fmaf(f1, c1[r], i1 * g1);
            h1f[r] = o1 * fast_tanh(c1[r]);
        }

        // rebuild hb = h^T f16 B-layout: pack pairs in place, no lane swap (own D rows = B's k)
#pragma unroll
        for (int v = 0; v < 4; ++v) {
            auto pk0 = __builtin_amdgcn_cvt_pkrtz(h0f[2 * v], h0f[2 * v + 1]);
            auto pk1 = __builtin_amdgcn_cvt_pkrtz(h1f[2 * v], h1f[2 * v + 1]);
            const int p0i = __builtin_bit_cast(int, pk0);
            const int p1i = __builtin_bit_cast(int, pk1);
            const int p0s = __builtin_amdgcn_ds_swizzle(p0i, SWZ_SWAPX16);
            const int p1s = __builtin_amdgcn_ds_swizzle(p1i, SWZ_SWAPX16);
            const v2h lo = __builtin_bit_cast(v2h, lh ? p0i : p0i); // K = 8*lh+2v,+1
            const v2h hi = __builtin_bit_cast(v2h, lh ? p1i : p1i); // K = 16+8*lh+2v,+1
            hb[2 * v]     = lo[0];
            hb[2 * v + 1] = lo[1];
            hb[8 + 2 * v] = hi[0];
            hb[9 + 2 * v] = hi[1];
        }
    }

    // ---- FC head: out[b] = fc_w[0:32].h_T + fc_w[32:36].mat_emb + fc_w[36:38].freq_emb + fc_b
    float acc = 0.0f;
#pragma unroll
    for (int r = 0; r < 8; ++r) {
        acc += h0f[r] * fc_w[8 * lh + r];        // m = r + 8*lh
        acc += h1f[r] * fc_w[16 + 8 * lh + r];   // m = 16 + r + 8*lh
    }
    const int ai = __builtin_bit_cast(int, acc);
    acc += __builtin_bit_cast(float, __builtin_amdgcn_ds_swizzle(ai, SWZ_SWAPX16));

    if (lh == 0) {
        const int b = b0 + r16;
        float res = acc + fc_b[0];
        const int mi = mat_idx[b];
        const int fi = freq_idx[b];
#pragma unroll
        for (int e = 0; e < 4; ++e) res += mat_emb[mi * 4 + e] * fc_w[32 + e];
#pragma unroll
        for (int e = 0; e < 2; ++e) res += freq_emb[fi * 2 + e] * fc_w[36 + e];
        out[b] = res;
    }
}

extern "C" void kernel_launch(void* const* d_in, const int* in_sizes, int n_in,
                              void* d_out, int out_size, void* d_ws, size_t ws_size,
                              hipStream_t stream) {
    const float* x        = (const float*)d_in[0];
    const int*   mat_idx  = (const int*)  d_in[1];
    const int*   freq_idx = (const int*)  d_in[2];
    const float* W_ih     = (const float*)d_in[3];
    const float* W_hh     = (const float*)d_in[4];
    const float* b_ih     = (const float*)d_in[5];
    const float* b_hh     = (const float*)d_in[6];
    const float* mat_emb  = (const float*)d_in[7];
    const float* freq_emb = (const float*)d_in[8];
    const float* fc_w     = (const float*)d_in[9];
    const float* fc_b     = (const float*)d_in[10];
    float* out = (float*)d_out;

    const int B = in_sizes[1];                 // mat_idx has B elements
    const int T = in_sizes[0] / (B * 3);       // x = [B, T, 3]

    const int waves  = (B + 15) / 16;          // one wave per 16 batch rows
    const int block  = 256;                    // 8 wave32 per workgroup
    const int grid   = (waves + 7) / 8;

    GlobalLSTM_kernel<<<grid, block, 0, stream>>>(
        x, mat_idx, freq_idx, W_ih, W_hh, b_ih, b_hh,
        mat_emb, freq_emb, fc_w, fc_b, out, B, T);
}
